// DiffusionConditionModel_31696858645159
// MI455X (gfx1250) — compile-verified, matrix operand order checked by reading, not run on hardware
//
#include <hip/hip_runtime.h>
#include <hip/hip_bf16.h>
#include <math.h>

// ---------------------------------------------------------------------------
// DiffusionConditionModel for MI455X (gfx1250, wave32, WMMA 16x16x32 f16)
//
// Model dims: B=4, N=128, HIDDEN=256, INPUT_DIM=256, PROTO=64, STEP_DIM=128,
// NUM_HEADS=8 (d=32), LAYER_NUM=2.
//
// d_out layout (floats): out[4*128*128*64] | attn_map[32*128*128*128] | pad_mask[4*128*128]
// d_ws needs >= 50,594,304 floats (~203 MB): X | bufQ | bufT | te | k | v
// ---------------------------------------------------------------------------

typedef _Float16 v16h __attribute__((ext_vector_type(16)));
typedef _Float16 v8h  __attribute__((ext_vector_type(8)));
typedef _Float16 v4h  __attribute__((ext_vector_type(4)));
typedef float    v8f  __attribute__((ext_vector_type(8)));
typedef float    v4f  __attribute__((ext_vector_type(4)));

#define ATT_INF 100000000.0f

// ---- WMMA helper -----------------------------------------------------------
__device__ __forceinline__ v8f wmma16(v16h a, v16h b, v8f c) {
    return __builtin_amdgcn_wmma_f32_16x16x32_f16(false, a, false, b, (short)0, c,
                                                  false, false);
}

// A/B fragment layout (ISA 7.12.2, 16-bit A 16x32):
//   lane l (<16): elements 0..7 = K[kbase+0..7],  8..15 = K[kbase+16..23]
//   lane l+16   : elements 0..7 = K[kbase+8..15], 8..15 = K[kbase+24..31]
// => per-lane: two 8-half chunks at kbase + 8*hi and kbase + 16 + 8*hi.
__device__ __forceinline__ v16h lds_load_frag(const _Float16* rowp, int kbase, int hi) {
    v8h lo = *(const v8h*)(rowp + kbase + hi * 8);
    v8h hh = *(const v8h*)(rowp + kbase + 16 + hi * 8);
    v16h r;
#pragma unroll
    for (int i = 0; i < 8; ++i) { r[i] = lo[i]; r[i + 8] = hh[i]; }
    return r;
}

__device__ __forceinline__ v16h gmem_load_frag_f32(const float* p, int hi) {
    v16h r;
    const float* p0 = p + hi * 8;
    const float* p1 = p + 16 + hi * 8;
#pragma unroll
    for (int i = 0; i < 8; i += 4) {
        v4f t0 = *(const v4f*)(p0 + i);
        v4f t1 = *(const v4f*)(p1 + i);
        r[i + 0] = (_Float16)t0.x; r[i + 1] = (_Float16)t0.y;
        r[i + 2] = (_Float16)t0.z; r[i + 3] = (_Float16)t0.w;
        r[i + 8] = (_Float16)t1.x; r[i + 9] = (_Float16)t1.y;
        r[i + 10] = (_Float16)t1.z; r[i + 11] = (_Float16)t1.w;
    }
    return r;
}

// ---------------------------------------------------------------------------
// Generic GEMM: C[M,N] = act(A[M,K] @ W[N,K]^T + bias[N]), f32 in/out,
// f16 WMMA inner product. Block tile 128x64, 256 threads (8 waves),
// wave micro-tile 32x32 (2x2 WMMA frags). K in {128,192,256}.
// ---------------------------------------------------------------------------
#define GEMM_BM 128
#define GEMM_BN 64

__launch_bounds__(256)
__global__ void gemm_wmma(const float* __restrict__ A, const float* __restrict__ W,
                          const float* __restrict__ bias, float* __restrict__ C,
                          int M, int N, int K, int do_relu)
{
    extern __shared__ char smem_raw[];
    const int KP = K + 16;                       // padded f16 stride (mult of 16)
    _Float16* As = (_Float16*)smem_raw;          // [128][KP]
    _Float16* Ws = As + GEMM_BM * KP;            // [64][KP]

    const int tid = threadIdx.x;
    const int m0 = blockIdx.x * GEMM_BM;
    const int n0 = blockIdx.y * GEMM_BN;

    // cooperative stage + f32->f16 convert
    {
        int total = GEMM_BM * K / 4;
        for (int i = tid; i < total; i += 256) {
            int e = i * 4; int r = e / K; int c = e - r * K;
            v4f t = *(const v4f*)(A + (size_t)(m0 + r) * K + c);
            v4h h; h[0] = (_Float16)t.x; h[1] = (_Float16)t.y;
            h[2] = (_Float16)t.z; h[3] = (_Float16)t.w;
            *(v4h*)(As + r * KP + c) = h;
        }
        total = GEMM_BN * K / 4;
        for (int i = tid; i < total; i += 256) {
            int e = i * 4; int r = e / K; int c = e - r * K;
            v4f t = *(const v4f*)(W + (size_t)(n0 + r) * K + c);
            v4h h; h[0] = (_Float16)t.x; h[1] = (_Float16)t.y;
            h[2] = (_Float16)t.z; h[3] = (_Float16)t.w;
            *(v4h*)(Ws + r * KP + c) = h;
        }
    }
    __syncthreads();

    const int wave = tid >> 5, lane = tid & 31;
    const int wm = (wave & 3) * 32;      // 4 waves along M
    const int wn = (wave >> 2) * 32;     // 2 waves along N
    const int lr = lane & 15, hi = lane >> 4;

    v8f acc[2][2] = {};
    for (int kc = 0; kc < K; kc += 32) {
        v16h a0 = lds_load_frag(As + (wm + lr) * KP, kc, hi);
        v16h a1 = lds_load_frag(As + (wm + 16 + lr) * KP, kc, hi);
        v16h b0 = lds_load_frag(Ws + (wn + lr) * KP, kc, hi);
        v16h b1 = lds_load_frag(Ws + (wn + 16 + lr) * KP, kc, hi);
        acc[0][0] = wmma16(a0, b0, acc[0][0]);
        acc[0][1] = wmma16(a0, b1, acc[0][1]);
        acc[1][0] = wmma16(a1, b0, acc[1][0]);
        acc[1][1] = wmma16(a1, b1, acc[1][1]);
    }

    // epilogue: C/D frag element (vgpr r, lane l) -> M = r + 8*(l/16), N = l%16
#pragma unroll
    for (int ti = 0; ti < 2; ++ti) {
#pragma unroll
        for (int tj = 0; tj < 2; ++tj) {
            int ncol = n0 + wn + tj * 16 + lr;
            float bv = bias[ncol];
#pragma unroll
            for (int r = 0; r < 8; ++r) {
                int mrow = m0 + wm + ti * 16 + r + 8 * hi;
                float v = acc[ti][tj][r] + bv;
                if (do_relu) v = fmaxf(v, 0.f);
                C[(size_t)mrow * N + ncol] = v;
            }
        }
    }
}

// ---------------------------------------------------------------------------
// Fused attention, one block per (b,h,i): scores (WMMA, K=d=32), masked
// softmax, attn@v (WMMA, K=128). Optionally writes softmaxed attention to
// attn_out (layer 2 -> d_out attn_map region, already head-major layout).
// Block = 128 threads (4 waves), each wave a 32-row j-slab.
// ---------------------------------------------------------------------------
#define KH_STRIDE 40
#define VT_STRIDE 136
#define S_STRIDE  130
#define P_STRIDE  136
// bytes: Kh 10240 | VhT 8704 | S 66560 | P 34816 | mb 512  = 120832
#define ATTN_LDS_BYTES (128*KH_STRIDE*2 + 32*VT_STRIDE*2 + 128*S_STRIDE*4 + 128*P_STRIDE*2 + 128*4)

__launch_bounds__(128)
__global__ void attn_fused(const float* __restrict__ q, const float* __restrict__ kbuf,
                           const float* __restrict__ vbuf, const int* __restrict__ inst,
                           float* __restrict__ out, float* __restrict__ attn_out)
{
    extern __shared__ char smem_raw[];
    _Float16* Kh  = (_Float16*)smem_raw;          // [128 k][KH_STRIDE] (uses 32 d)
    _Float16* VhT = Kh + 128 * KH_STRIDE;         // [32 d][VT_STRIDE] (uses 128 k)
    float*    S   = (float*)(VhT + 32 * VT_STRIDE); // [128 j][S_STRIDE]
    _Float16* P   = (_Float16*)(S + 128 * S_STRIDE); // [128 j][P_STRIDE]
    float*    mb  = (float*)(P + 128 * P_STRIDE);    // [128] mask bias

    const int i = blockIdx.x, h = blockIdx.y, b = blockIdx.z;
    const int tid = threadIdx.x, lane = tid & 31, wave = tid >> 5;
    const int lr = lane & 15, hi = lane >> 4;
    const int ninst = inst[b];

    // ---- stage K-head (row-major) and V-head (transposed) as f16 ----------
    {
        const float* kp = kbuf + ((size_t)(b * 128 + tid)) * 256 + h * 32;
        const float* vp = vbuf + ((size_t)(b * 128 + tid)) * 256 + h * 32;
#pragma unroll
        for (int d = 0; d < 32; d += 4) {
            v4f kt = *(const v4f*)(kp + d);
            v4h kh; kh[0] = (_Float16)kt.x; kh[1] = (_Float16)kt.y;
            kh[2] = (_Float16)kt.z; kh[3] = (_Float16)kt.w;
            *(v4h*)(Kh + tid * KH_STRIDE + d) = kh;
            v4f vt = *(const v4f*)(vp + d);
            VhT[(d + 0) * VT_STRIDE + tid] = (_Float16)vt.x;
            VhT[(d + 1) * VT_STRIDE + tid] = (_Float16)vt.y;
            VhT[(d + 2) * VT_STRIDE + tid] = (_Float16)vt.z;
            VhT[(d + 3) * VT_STRIDE + tid] = (_Float16)vt.w;
        }
        mb[tid] = (tid < ninst) ? 0.f : -ATT_INF;
    }
    __syncthreads();

    // ---- scores: per wave 32 j-rows x 128 k-cols, K=d=32 (1 WMMA/tile) ----
    const size_t qrow0 = (size_t)((b * 128 + i) * 128) + wave * 32;
    v16h a0 = gmem_load_frag_f32(q + (qrow0 + lr) * 256 + h * 32, hi);
    v16h a1 = gmem_load_frag_f32(q + (qrow0 + 16 + lr) * 256 + h * 32, hi);
    const float scale = 0.17677669529663687f;   // 1/sqrt(32)
#pragma unroll
    for (int nt = 0; nt < 8; ++nt) {
        const int k0 = nt * 16;
        v16h bk = lds_load_frag(Kh + (k0 + lr) * KH_STRIDE, 0, hi);
        v8f c0 = {}; v8f c1 = {};
        c0 = wmma16(a0, bk, c0);
        c1 = wmma16(a1, bk, c1);
#pragma unroll
        for (int r = 0; r < 8; ++r) {
            S[(wave * 32 + r + 8 * hi) * S_STRIDE + k0 + lr] = c0[r] * scale;
            S[(wave * 32 + 16 + r + 8 * hi) * S_STRIDE + k0 + lr] = c1[r] * scale;
        }
    }
    __syncthreads();

    // ---- masked softmax: thread tid owns row j = tid ----------------------
    {
        float* srow = S + tid * S_STRIDE;
        float m = -3.4e38f;
        for (int k2 = 0; k2 < 128; ++k2) {
            float x = srow[k2] + mb[k2];
            srow[k2] = x;
            m = fmaxf(m, x);
        }
        float s = 0.f;
        for (int k2 = 0; k2 < 128; ++k2) {
            float e = __expf(srow[k2] - m);
            srow[k2] = e;
            s += e;
        }
        float inv = 1.f / s;
        _Float16* prow = P + tid * P_STRIDE;
        for (int k2 = 0; k2 < 128; ++k2) {
            float pv = srow[k2] * inv;
            srow[k2] = pv;
            prow[k2] = (_Float16)pv;
        }
    }
    __syncthreads();

    // ---- emit attn_map (layer 2): [h*B+b][i][j][k], coalesced over k ------
    if (attn_out) {
        size_t base = ((((size_t)h * 4 + b) * 128 + i) * 128) * 128;
        for (int j = 0; j < 128; ++j)
            attn_out[base + (size_t)j * 128 + tid] = S[j * S_STRIDE + tid];
    }

    // ---- attn@v: M=32 j, N=32 d, K=128 ------------------------------------
    v8f acc[2][2] = {};
#pragma unroll
    for (int kc = 0; kc < 128; kc += 32) {
        v16h p0 = lds_load_frag(P + (wave * 32 + lr) * P_STRIDE, kc, hi);
        v16h p1 = lds_load_frag(P + (wave * 32 + 16 + lr) * P_STRIDE, kc, hi);
        v16h v0 = lds_load_frag(VhT + lr * VT_STRIDE, kc, hi);
        v16h v1 = lds_load_frag(VhT + (16 + lr) * VT_STRIDE, kc, hi);
        acc[0][0] = wmma16(p0, v0, acc[0][0]);
        acc[0][1] = wmma16(p0, v1, acc[0][1]);
        acc[1][0] = wmma16(p1, v0, acc[1][0]);
        acc[1][1] = wmma16(p1, v1, acc[1][1]);
    }
#pragma unroll
    for (int ti = 0; ti < 2; ++ti) {
#pragma unroll
        for (int tj = 0; tj < 2; ++tj) {
#pragma unroll
            for (int r = 0; r < 8; ++r) {
                size_t p = qrow0 + ti * 16 + r + 8 * hi;
                out[p * 256 + h * 32 + tj * 16 + lr] = acc[ti][tj][r];
            }
        }
    }
}

// ---------------------------------------------------------------------------
// out = LayerNorm(x + res) * pad_mask   (256 channels, one wave per position)
// ---------------------------------------------------------------------------
__launch_bounds__(256)
__global__ void ln_res_mask(const float* __restrict__ x, const float* __restrict__ res,
                            const float* __restrict__ g, const float* __restrict__ be,
                            const int* __restrict__ inst, float* __restrict__ out)
{
    const int wave = threadIdx.x >> 5, lane = threadIdx.x & 31;
    const size_t p = (size_t)blockIdx.x * 8 + wave;     // < 65536
    const int b = (int)(p >> 14);
    const int ij = (int)(p & 16383);
    const int i = ij >> 7, j = ij & 127;
    const int n = inst[b];
    const float npm = (i < n && j < n) ? 1.f : 0.f;

    const float* xp = x + p * 256;
    const float* rp = res + p * 256;
    float v[8];
    float s = 0.f;
#pragma unroll
    for (int u = 0; u < 8; ++u) {
        float t = xp[lane + 32 * u] + rp[lane + 32 * u];
        v[u] = t; s += t;
    }
#pragma unroll
    for (int o = 16; o > 0; o >>= 1) s += __shfl_xor(s, o, 32);
    const float mean = s * (1.f / 256.f);
    float vs = 0.f;
#pragma unroll
    for (int u = 0; u < 8; ++u) { float d = v[u] - mean; vs += d * d; }
#pragma unroll
    for (int o = 16; o > 0; o >>= 1) vs += __shfl_xor(vs, o, 32);
    const float rstd = rsqrtf(vs * (1.f / 256.f) + 1e-5f);
#pragma unroll
    for (int u = 0; u < 8; ++u) {
        int c = lane + 32 * u;
        out[p * 256 + c] = ((v[u] - mean) * rstd * g[c] + be[c]) * npm;
    }
}

// ---------------------------------------------------------------------------
// Timestep embedding + se linear: te[b,:] = [cos(t*f)|sin(t*f)] @ se_w^T + se_b
// ---------------------------------------------------------------------------
__launch_bounds__(128)
__global__ void te_kernel(const float* __restrict__ tsteps, const float* __restrict__ se_w,
                          const float* __restrict__ se_b, float* __restrict__ te)
{
    __shared__ float emb[128];
    const int t = threadIdx.x;
    const float lnmp = logf(10000.f);
    for (int b = 0; b < 4; ++b) {
        if (t < 64) {
            float freq = __expf(-lnmp * (float)t / 64.f);
            float arg = tsteps[b] * freq;
            emb[t] = __cosf(arg);
            emb[t + 64] = __sinf(arg);
        }
        __syncthreads();
        float s = se_b[t];
        for (int c = 0; c < 128; ++c) s += emb[c] * se_w[t * 128 + c];
        te[b * 128 + t] = s;
        __syncthreads();
    }
}

// Build fusion input: Xin[p, 0:64] = sgs_t[p], Xin[p, 64:192] = te[b]
__launch_bounds__(192)
__global__ void build_fusion_in(const float* __restrict__ sgs, const float* __restrict__ te,
                                float* __restrict__ xin)
{
    const size_t p = blockIdx.x;
    const int t = threadIdx.x;
    const int b = (int)(p >> 14);
    float v = (t < 64) ? sgs[p * 64 + t] : te[b * 128 + (t - 64)];
    xin[p * 192 + t] = v;
}

__launch_bounds__(256)
__global__ void pad_mask_kernel(const int* __restrict__ inst, float* __restrict__ out)
{
    const size_t p = (size_t)blockIdx.x * 256 + threadIdx.x;   // 65536 total
    const int b = (int)(p >> 14);
    const int ij = (int)(p & 16383);
    const int i = ij >> 7, j = ij & 127;
    const int n = inst[b];
    out[p] = (i < n && j < n) ? 1.f : 0.f;
}

// ---------------------------------------------------------------------------
extern "C" void kernel_launch(void* const* d_in, const int* in_sizes, int n_in,
                              void* d_out, int out_size, void* d_ws, size_t ws_size,
                              hipStream_t stream)
{
    const float* sgs   = (const float*)d_in[0];
    const float* tstep = (const float*)d_in[1];
    const float* pf    = (const float*)d_in[2];
    const int*   inst  = (const int*)  d_in[3];
    const float* se_w  = (const float*)d_in[4];
    const float* se_b  = (const float*)d_in[5];
    const float* fu_w1 = (const float*)d_in[6];
    const float* fu_b1 = (const float*)d_in[7];
    const float* fu_w2 = (const float*)d_in[8];
    const float* fu_b2 = (const float*)d_in[9];
    const float* cs_w  = (const float*)d_in[10];
    const float* cs_b  = (const float*)d_in[11];
    const float* wq_w  = (const float*)d_in[12];
    const float* wq_b  = (const float*)d_in[13];
    const float* wk_w  = (const float*)d_in[14];
    const float* wk_b  = (const float*)d_in[15];
    const float* wv_w  = (const float*)d_in[16];
    const float* wv_b  = (const float*)d_in[17];
    const float* fc_w  = (const float*)d_in[18];
    const float* fc_b  = (const float*)d_in[19];
    const float* c1_w  = (const float*)d_in[20];
    const float* c1_b  = (const float*)d_in[21];
    const float* c2_w  = (const float*)d_in[22];
    const float* c2_b  = (const float*)d_in[23];
    const float* ln1_g = (const float*)d_in[24];
    const float* ln1_b = (const float*)d_in[25];
    const float* ln2_g = (const float*)d_in[26];
    const float* ln2_b = (const float*)d_in[27];

    float* outp = (float*)d_out;
    float* out_main = outp;                       // [65536, 64]
    float* out_attn = outp + 4194304;             // [32,128,128,128]
    float* out_pad  = outp + 71303168;            // [65536]

    float* ws   = (float*)d_ws;
    float* X    = ws;                   // [65536,256]
    float* bufQ = X + 16777216;         // [65536,256]
    float* bufT = bufQ + 16777216;      // [65536,256] (also holds Xin [65536,192])
    float* te   = bufT + 16777216;      // [4,128]
    float* kb   = te + 512;             // [512,256]
    float* vb   = kb + 131072;          // [512,256]

    auto gemm_lds = [](int K) { return (size_t)((GEMM_BM + GEMM_BN) * (K + 16) * 2); };

    // timestep embedding + step linear
    te_kernel<<<1, 128, 0, stream>>>(tstep, se_w, se_b, te);
    // fusion input concat
    build_fusion_in<<<65536, 192, 0, stream>>>(sgs, te, bufT);
    // fusion MLP
    gemm_wmma<<<dim3(512, 4), 256, gemm_lds(192), stream>>>(bufT, fu_w1, fu_b1, bufQ,
                                                            65536, 256, 192, 1);
    gemm_wmma<<<dim3(512, 4), 256, gemm_lds(256), stream>>>(bufQ, fu_w2, fu_b2, X,
                                                            65536, 256, 256, 0);

    for (int l = 0; l < 2; ++l) {
        const size_t wo = (size_t)l * 65536, bo = (size_t)l * 256;
        // k, v projections of proposal features [512,256]
        gemm_wmma<<<dim3(4, 4), 256, gemm_lds(256), stream>>>(pf, wk_w + wo, wk_b + bo, kb,
                                                              512, 256, 256, 0);
        gemm_wmma<<<dim3(4, 4), 256, gemm_lds(256), stream>>>(pf, wv_w + wo, wv_b + bo, vb,
                                                              512, 256, 256, 0);
        // q projection [65536,256]
        gemm_wmma<<<dim3(512, 4), 256, gemm_lds(256), stream>>>(X, wq_w + wo, wq_b + bo, bufQ,
                                                                65536, 256, 256, 0);
        // fused attention (layer 2 also emits attn_map)
        attn_fused<<<dim3(128, 8, 4), 128, ATTN_LDS_BYTES, stream>>>(
            bufQ, kb, vb, inst, bufT, (l == 1) ? out_attn : nullptr);
        // fc + residual LN + mask
        gemm_wmma<<<dim3(512, 4), 256, gemm_lds(256), stream>>>(bufT, fc_w + wo, fc_b + bo, bufQ,
                                                                65536, 256, 256, 0);
        ln_res_mask<<<8192, 256, 0, stream>>>(bufQ, X, ln1_g + bo, ln1_b + bo, inst, X);
        // FFN (conv1x1) + residual LN + mask
        gemm_wmma<<<dim3(512, 4), 256, gemm_lds(256), stream>>>(X, c1_w + wo, c1_b + bo, bufT,
                                                                65536, 256, 256, 1);
        gemm_wmma<<<dim3(512, 4), 256, gemm_lds(256), stream>>>(bufT, c2_w + wo, c2_b + bo, bufQ,
                                                                65536, 256, 256, 0);
        ln_res_mask<<<8192, 256, 0, stream>>>(bufQ, X, ln2_g + bo, ln2_b + bo, inst, X);
    }

    // compress_state -> main output [65536,64]
    gemm_wmma<<<dim3(512, 1), 256, gemm_lds(256), stream>>>(X, cs_w, cs_b, out_main,
                                                            65536, 64, 256, 0);
    // pad mask output
    pad_mask_kernel<<<256, 256, 0, stream>>>(inst, out_pad);
}
